// GSL_32255204393055
// MI455X (gfx1250) — matrix operand order repaired and checked by reading, not compiled, not run on hardware
//
#include <hip/hip_runtime.h>
#include <hip/hip_bf16.h>

#define NROWS    10000
#define DIM      256
#define NTILES16 625          // valid 16-row tiles (10000/16)
#define NPADT    626          // padded 16-row tiles
#define NSUPER   313          // 32-row supertiles (10016/32)

typedef __attribute__((ext_vector_type(16))) __bf16 v16bf;
typedef __attribute__((ext_vector_type(8)))  float  v8f;

union ABfrag { v16bf v; uint4 q[2]; };

// ---------------------------------------------------------------------------
// Kernel 1: fused MLP + L2-normalize + bf16 hi/lo split, with zero-padding of
// rows [10000, 10016) so the WMMA kernel can use uniform 32-row supertiles.
// ---------------------------------------------------------------------------
__global__ void mlp_norm_kernel(const float* __restrict__ h,
                                const float* __restrict__ W1,
                                const float* __restrict__ b1,
                                const float* __restrict__ W2,
                                const float* __restrict__ b2,
                                __hip_bfloat16* __restrict__ ehi,
                                __hip_bfloat16* __restrict__ elo)
{
    __shared__ float bufA[16 * DIM];
    __shared__ float bufB[16 * DIM];
    __shared__ float rnorm[16];

    const int tid     = threadIdx.x;
    const int rowBase = blockIdx.x * 16;

    if (rowBase >= NROWS) {                       // pad rows -> zeros
        const __hip_bfloat16 z = __float2bfloat16(0.0f);
        #pragma unroll
        for (int r = 0; r < 16; ++r) {
            size_t idx = (size_t)(rowBase + r) * DIM + tid;
            ehi[idx] = z;
            elo[idx] = z;
        }
        return;
    }

    for (int i = tid; i < 16 * DIM; i += 256)
        bufA[i] = h[(size_t)rowBase * DIM + i];
    __syncthreads();

    float acc[16];
    // ---- layer 1: x1 = relu(h @ W1^T + b1) ----
    #pragma unroll
    for (int r = 0; r < 16; ++r) acc[r] = b1[tid];
    {
        const float* w = W1 + (size_t)tid * DIM;
        for (int d = 0; d < DIM; ++d) {
            float wv = w[d];
            #pragma unroll
            for (int r = 0; r < 16; ++r)
                acc[r] = fmaf(bufA[r * DIM + d], wv, acc[r]);
        }
    }
    #pragma unroll
    for (int r = 0; r < 16; ++r) bufB[r * DIM + tid] = fmaxf(acc[r], 0.0f);
    __syncthreads();

    // ---- layer 2: x2 = x1 @ W2^T + b2 ----
    #pragma unroll
    for (int r = 0; r < 16; ++r) acc[r] = b2[tid];
    {
        const float* w = W2 + (size_t)tid * DIM;
        for (int d = 0; d < DIM; ++d) {
            float wv = w[d];
            #pragma unroll
            for (int r = 0; r < 16; ++r)
                acc[r] = fmaf(bufB[r * DIM + d], wv, acc[r]);
        }
    }
    #pragma unroll
    for (int r = 0; r < 16; ++r) bufA[r * DIM + tid] = acc[r];
    __syncthreads();

    // ---- per-row L2 norm ----
    if (tid < 16) {
        float s = 0.0f;
        for (int j = 0; j < DIM; ++j) {
            float v = bufA[tid * DIM + j];
            s = fmaf(v, v, s);
        }
        rnorm[tid] = 1.0f / fmaxf(sqrtf(s), 1e-12f);
    }
    __syncthreads();

    // ---- normalize + bf16 hi/lo split (error-compensated GEMM inputs) ----
    #pragma unroll
    for (int r = 0; r < 16; ++r) {
        float e = bufA[r * DIM + tid] * rnorm[r];
        __hip_bfloat16 hi = __float2bfloat16(e);
        float lo_f = e - __bfloat162float(hi);
        size_t idx = (size_t)(rowBase + r) * DIM + tid;
        ehi[idx] = hi;
        elo[idx] = __float2bfloat16(lo_f);
    }
}

// ---------------------------------------------------------------------------
// Kernel 2: adj = e @ e^T via v_wmma_f32_16x16x32_bf16, error-compensated
// (hi*hi + hi*lo + lo*hi). One wave computes a 32x32 region (2x2 tiles,
// halved fragment traffic); only the upper triangle of supertiles is
// computed, with the mirror image stored as contiguous per-lane b128 writes.
// ---------------------------------------------------------------------------
__device__ __forceinline__ v8f wmma3(v16bf ah, v16bf al, v16bf bh, v16bf bl, v8f c)
{
    c = __builtin_amdgcn_wmma_f32_16x16x32_bf16(false, ah, false, bh, (short)0, c, false, false);
    c = __builtin_amdgcn_wmma_f32_16x16x32_bf16(false, ah, false, bl, (short)0, c, false, false);
    c = __builtin_amdgcn_wmma_f32_16x16x32_bf16(false, al, false, bh, (short)0, c, false, false);
    return c;
}

__global__ void adj_wmma_kernel(const __hip_bfloat16* __restrict__ ehi,
                                const __hip_bfloat16* __restrict__ elo,
                                float* __restrict__ out)
{
    const int lane   = threadIdx.x & 31;
    const int wave   = threadIdx.x >> 5;
    const int superR = blockIdx.y * 2 + (wave >> 2);
    const int superC = blockIdx.x * 4 + (wave & 3);
    if (superR >= NSUPER || superC >= NSUPER || superR > superC)
        return;                                   // wave-uniform exit

    const int half = lane >> 4;                   // 0: lanes 0-15, 1: lanes 16-31
    const int mn   = lane & 15;

    const size_t ar0 = (size_t)(superR * 32 +      mn) * DIM;   // A row tile 0
    const size_t ar1 = (size_t)(superR * 32 + 16 + mn) * DIM;   // A row tile 1
    const size_t br0 = (size_t)(superC * 32 +      mn) * DIM;   // B col tile 0
    const size_t br1 = (size_t)(superC * 32 + 16 + mn) * DIM;   // B col tile 1

    v8f acc[2][2] = {{{}, {}}, {{}, {}}};

    #pragma unroll
    for (int kk = 0; kk < DIM; kk += 32) {
        ABfrag a0h, a0l, a1h, a1l, b0h, b0l, b1h, b1l;
        const int aoff = kk + 16 * half;          // A: 16 contiguous halfs
        a0h.q[0] = *(const uint4*)(ehi + ar0 + aoff);
        a0h.q[1] = *(const uint4*)(ehi + ar0 + aoff + 8);
        a0l.q[0] = *(const uint4*)(elo + ar0 + aoff);
        a0l.q[1] = *(const uint4*)(elo + ar0 + aoff + 8);
        a1h.q[0] = *(const uint4*)(ehi + ar1 + aoff);
        a1h.q[1] = *(const uint4*)(ehi + ar1 + aoff + 8);
        a1l.q[0] = *(const uint4*)(elo + ar1 + aoff);
        a1l.q[1] = *(const uint4*)(elo + ar1 + aoff + 8);
        const int boff = kk + 16 * half;          // B: 16 contiguous halfs
        b0h.q[0] = *(const uint4*)(ehi + br0 + boff);
        b0h.q[1] = *(const uint4*)(ehi + br0 + boff + 8);
        b0l.q[0] = *(const uint4*)(elo + br0 + boff);
        b0l.q[1] = *(const uint4*)(elo + br0 + boff + 8);
        b1h.q[0] = *(const uint4*)(ehi + br1 + boff);
        b1h.q[1] = *(const uint4*)(ehi + br1 + boff + 8);
        b1l.q[0] = *(const uint4*)(elo + br1 + boff);
        b1l.q[1] = *(const uint4*)(elo + br1 + boff + 8);

        acc[0][0] = wmma3(a0h.v, a0l.v, b0h.v, b0l.v, acc[0][0]);
        acc[0][1] = wmma3(a0h.v, a0l.v, b1h.v, b1l.v, acc[0][1]);
        acc[1][0] = wmma3(a1h.v, a1l.v, b0h.v, b0l.v, acc[1][0]);
        acc[1][1] = wmma3(a1h.v, a1l.v, b1h.v, b1l.v, acc[1][1]);
    }

    // ---- stores: direct (upper triangle) + mirrored (lower triangle) ----
    #pragma unroll
    for (int i = 0; i < 2; ++i) {
        const int rt = superR * 2 + i;            // 16-row tile index
        if (rt >= NTILES16) continue;
        #pragma unroll
        for (int j = 0; j < 2; ++j) {
            const int ct = superC * 2 + j;        // 16-col tile index
            if (ct >= NTILES16) continue;
            const v8f c = acc[i][j];
            // direct: VGPR r -> row (rt*16 + r + 8*half), col (ct*16 + mn)
            float* p = out + (size_t)(rt * 16 + 8 * half) * NROWS + ct * 16 + mn;
            #pragma unroll
            for (int r = 0; r < 8; ++r)
                p[(size_t)r * NROWS] = c[r];
            // mirror: row (ct*16 + mn), cols (rt*16 + 8*half + r) contiguous
            if (superR != superC) {
                float* q = out + (size_t)(ct * 16 + mn) * NROWS + rt * 16 + 8 * half;
                #pragma unroll
                for (int r = 0; r < 8; ++r)
                    q[r] = c[r];
            }
        }
    }
}

// ---------------------------------------------------------------------------
// Kernel 3: per-row top-(k+1) threshold via 4-pass byte radix-select on
// order-preserving float keys, then in-place mask + relu. One block per row.
// ---------------------------------------------------------------------------
__device__ __forceinline__ unsigned fkey(float f)
{
    unsigned u = __float_as_uint(f);
    return (u & 0x80000000u) ? ~u : (u | 0x80000000u);
}

__global__ void topk_mask_kernel(const int* __restrict__ kptr,
                                 float* __restrict__ adj, int n)
{
    __shared__ unsigned hist[256];
    __shared__ unsigned s_prefix;
    __shared__ int      s_k;

    const int tid = threadIdx.x;
    const int row = blockIdx.x;
    float* rowp = adj + (size_t)row * n;

    if (tid == 0) { s_prefix = 0u; s_k = kptr[0] + 1; }
    __syncthreads();

    for (int p = 3; p >= 0; --p) {
        hist[tid] = 0u;
        __syncthreads();
        const unsigned prefix  = s_prefix;
        const unsigned mask_hi = (p == 3) ? 0u : (0xFFFFFFFFu << (8 * (p + 1)));
        for (int j = tid; j < n; j += 256) {
            unsigned key = fkey(rowp[j]);
            if ((key & mask_hi) == (prefix & mask_hi))
                atomicAdd(&hist[(key >> (8 * p)) & 255u], 1u);
        }
        __syncthreads();
        if (tid == 0) {
            int k = s_k;
            unsigned b = 255u;
            for (;;) {
                int c = (int)hist[b];
                if (c >= k || b == 0u) break;
                k -= c;
                --b;
            }
            s_prefix = prefix | (b << (8 * p));
            s_k = k;
        }
        __syncthreads();
    }

    const unsigned tkey = s_prefix;               // key of (k+1)-th largest
    for (int j = tid; j < n; j += 256) {
        float v = rowp[j];
        rowp[j] = (fkey(v) >= tkey) ? fmaxf(v, 0.0f) : 0.0f;
    }
}

// ---------------------------------------------------------------------------
extern "C" void kernel_launch(void* const* d_in, const int* in_sizes, int n_in,
                              void* d_out, int out_size, void* d_ws, size_t ws_size,
                              hipStream_t stream)
{
    (void)in_sizes; (void)n_in; (void)out_size; (void)ws_size;

    const float* h  = (const float*)d_in[0];
    const float* W1 = (const float*)d_in[1];
    const float* b1 = (const float*)d_in[2];
    const float* W2 = (const float*)d_in[3];
    const float* b2 = (const float*)d_in[4];
    const int*   kp = (const int*)d_in[5];
    float* out = (float*)d_out;

    __hip_bfloat16* ehi = (__hip_bfloat16*)d_ws;                  // 10016*256 bf16
    __hip_bfloat16* elo = ehi + (size_t)NPADT * 16 * DIM;         // 10016*256 bf16

    mlp_norm_kernel<<<NPADT, 256, 0, stream>>>(h, W1, b1, W2, b2, ehi, elo);

    // 8 waves/block arranged 2x4 over 32x32 supertiles
    dim3 grid((NSUPER + 3) / 4, (NSUPER + 1) / 2, 1);
    adj_wmma_kernel<<<grid, 256, 0, stream>>>(ehi, elo, out);

    topk_mask_kernel<<<NROWS, 256, 0, stream>>>(kp, out, NROWS);
}
